// FCudaFramework_32908039421981
// MI455X (gfx1250) — matrix operand order repaired and checked by reading, not compiled, not run on hardware
//
#include <hip/hip_runtime.h>

// ---------------------------------------------------------------------------
// MI455X (gfx1250) implementation of SparseAttention + LocalizedRefinement +
// TemporalFeaturePropagation + FastCoverageConv.
// B=4, C=64, H=W=64, N=4096, top-k keep = int(4096*0.8) = 3276.
// ---------------------------------------------------------------------------

typedef __bf16 bf16;
typedef __attribute__((ext_vector_type(16))) __bf16 v16bf;
typedef __attribute__((ext_vector_type(8)))  __bf16 v8bf;
typedef __attribute__((ext_vector_type(8)))  float  v8f;
typedef __attribute__((ext_vector_type(4)))  float  v4f;

#define NB 4
#define NC 64
#define NH 64
#define NW 64
#define NN 4096            // tokens per batch
#define KKEEP 3276u        // int(4096 * 0.8)
#define RS 4104            // padded score row stride (f32), 16416B, 16B-aligned

// ---------------------------------------------------------------------------
// Kernel 1: QKV projection.  qkv[b,n,o] = sum_c x[b,c,n] * w_qkv[o,c].
// Q,K stored token-major bf16 [B][N][64]; V stored channel-major bf16
// [B][64][N] so that every WMMA B-fragment later is a contiguous 32B load.
// ---------------------------------------------------------------------------
__global__ __launch_bounds__(256) void qkv_kernel(
    const float* __restrict__ x, const float* __restrict__ wqkv,
    bf16* __restrict__ Qb, bf16* __restrict__ Kb, bf16* __restrict__ Vt) {
  __shared__ float xt[64 * 64];        // [c'][i] row-major, conflict-free
  __shared__ float wl[192 * 64];       // staged w_qkv
  const int b  = blockIdx.x >> 6;
  const int n0 = (blockIdx.x & 63) << 6;
  const float* xb = x + (size_t)b * NC * NN;

  for (int idx = threadIdx.x; idx < 64 * 64; idx += 256) {
    int cp = idx >> 6, i = idx & 63;
    xt[cp * 64 + i] = xb[(size_t)cp * NN + n0 + i];   // coalesced along n
  }
  for (int idx = threadIdx.x; idx < 192 * 64; idx += 256) wl[idx] = wqkv[idx];
  __syncthreads();

  const int i  = threadIdx.x & 63;     // token within tile
  const int og = threadIdx.x >> 6;     // output group 0..3 (48 outputs each)
  const int n  = n0 + i;
  for (int o = og * 48; o < og * 48 + 48; ++o) {
    float acc = 0.0f;
    for (int cp = 0; cp < 64; ++cp)
      acc = fmaf(xt[cp * 64 + i], wl[o * 64 + cp], acc);
    bf16 v = (bf16)acc;
    if (o < 64)        Qb[((size_t)(b * NN + n)) * 64 + o]        = v;
    else if (o < 128)  Kb[((size_t)(b * NN + n)) * 64 + (o - 64)] = v;
    else               Vt[((size_t)b * 64 + (o - 128)) * NN + n]  = v;
  }
}

// ---------------------------------------------------------------------------
// Kernel 2: attention.  One block = 16 query rows of one batch.
// 512 threads = 16 waves. Scores tile (16 x 4096 f32) lives in LDS
// (256KB of the 320KB WGP LDS). Exact per-row top-k via 4-pass radix select
// on monotonic float keys, masked softmax (unnormalized), P.V via bf16 WMMA.
// ---------------------------------------------------------------------------
__global__ __launch_bounds__(512, 1) void attn_kernel(
    const bf16* __restrict__ Qb, const bf16* __restrict__ Kb,
    const bf16* __restrict__ Vt, float* __restrict__ xcT) {
  extern __shared__ char smem[];
  float*    scores = (float*)smem;                                      // 16*RS
  unsigned* hist   = (unsigned*)(smem + 16 * RS * 4);                   // 16*256
  bf16*     qtile  = (bf16*)(smem + 16 * RS * 4 + 16 * 256 * 4);        // 16*80
  float*    outAcc = (float*)(smem + 16 * RS * 4 + 16 * 256 * 4 + 16 * 80 * 2);
  float*    rowsum = outAcc + 16 * 64;                                  // 16

  const int b    = blockIdx.x >> 8;
  const int n0   = (blockIdx.x & 255) << 4;
  const int tid  = threadIdx.x;
  const int lane = tid & 31;
  const int wv   = tid >> 5;            // wave 0..15
  const int l16  = lane & 15;
  const int kbA  = (lane < 16) ? 0 : 16;  // A fragment: K {0..15}/{16..31}
  const int kbB  = (lane < 16) ? 0 : 16;  // B fragment: K {0..15}/{16..31}

  // Stage Q tile (16x64 bf16, row stride 80 -> 160B, 16B aligned rows)
  for (int idx = tid; idx < 16 * 64; idx += 512) {
    int r = idx >> 6, c = idx & 63;
    qtile[r * 80 + c] = Qb[((size_t)(b * NN + n0 + r)) * 64 + c];
  }
  for (int idx = tid; idx < 16 * 64; idx += 512) outAcc[idx] = 0.0f;
  __syncthreads();

  // Per-wave A fragments of the Q tile (K chunk 0..31 and 32..63)
  v16bf a0, a1;
  {
    const bf16* qr = qtile + l16 * 80;
    v8bf q0 = *(const v8bf*)(qr + kbA);
    v8bf q1 = *(const v8bf*)(qr + kbA + 8);
    v8bf q2 = *(const v8bf*)(qr + 32 + kbA);
    v8bf q3 = *(const v8bf*)(qr + 32 + kbA + 8);
#pragma unroll
    for (int t = 0; t < 8; ++t) {
      a0[t] = q0[t]; a0[t + 8] = q1[t];
      a1[t] = q2[t]; a1[t + 8] = q3[t];
    }
  }

  // ---- Phase S: scores = Q K^T / 8.  Wave wv owns m-tiles [wv*16, wv*16+16)
  const bf16* Kbase = Kb + (size_t)b * NN * 64;
  for (int mt = wv * 16; mt < wv * 16 + 16; ++mt) {
    const bf16* kr = Kbase + (size_t)(mt * 16 + l16) * 64;
    v16bf b0 = *(const v16bf*)(kr + kbB);        // K rows are B^T rows
    v16bf b1 = *(const v16bf*)(kr + 32 + kbB);
    v8f acc = {};
    acc = __builtin_amdgcn_wmma_f32_16x16x32_bf16(false, a0, false, b0,
                                                  (short)0, acc, false, false);
    acc = __builtin_amdgcn_wmma_f32_16x16x32_bf16(false, a1, false, b1,
                                                  (short)0, acc, false, false);
    const int col = mt * 16 + l16;
    const int rb  = (lane >> 4) << 3;            // D rows i / i+8 per half-wave
#pragma unroll
    for (int i = 0; i < 8; ++i)
      scores[(rb + i) * RS + col] = acc[i] * 0.125f;   // 1/sqrt(64)
  }
  __syncthreads();

  // ---- Phase T: per-row top-k threshold + softmax.  Wave wv <-> row wv.
  const int r = wv;
  float* srow = scores + r * RS;
  float mx = -3.0e38f;
  for (int j = lane; j < NN; j += 32) mx = fmaxf(mx, srow[j]);
#pragma unroll
  for (int off = 16; off; off >>= 1) mx = fmaxf(mx, __shfl_xor(mx, off));

  unsigned* h = hist + r * 256;
  unsigned prefix = 0;
  unsigned want = KKEEP;                 // find the 3276-th largest key
  for (int pass = 0; pass < 4; ++pass) {
    const int shift = 24 - 8 * pass;
    for (int t = lane; t < 256; t += 32) h[t] = 0u;
    __syncthreads();
    for (int j = lane; j < NN; j += 32) {
      unsigned u   = __float_as_uint(srow[j]);
      unsigned key = u ^ ((unsigned)((int)u >> 31) | 0x80000000u);
      bool match = (pass == 0) || ((key >> (shift + 8)) == (prefix >> (shift + 8)));
      if (match) atomicAdd(&h[(key >> shift) & 255u], 1u);
    }
    __syncthreads();
    if (lane == 0) {
      unsigned cum = 0; int bin = 0;
      for (int t = 255; t >= 0; --t) {
        unsigned c = h[t];
        if (cum + c >= want) { bin = t; want -= cum; break; }
        cum += c;
      }
      prefix |= ((unsigned)bin) << shift;
    }
    prefix = (unsigned)__shfl((int)prefix, 0);
    want   = (unsigned)__shfl((int)want, 0);
    __syncthreads();
  }
  const unsigned tkey = prefix;          // exact key of k-th largest score

  float sum = 0.0f;
  for (int j = lane; j < NN; j += 32) {
    float s = srow[j];
    unsigned u   = __float_as_uint(s);
    unsigned key = u ^ ((unsigned)((int)u >> 31) | 0x80000000u);
    float p = (key >= tkey) ? __expf(s - mx) : 0.0f;   // masked = exp(-1e10)->0
    srow[j] = p;                         // unnormalized; fold 1/sum at the end
    sum += p;
  }
#pragma unroll
  for (int off = 16; off; off >>= 1) sum += __shfl_xor(sum, off);
  if (lane == 0) rowsum[r] = sum;
  __syncthreads();

  // ---- Phase PV: out = P . V.  Wave -> (c-tile = wv&3, m-quarter = wv>>2)
  const int c0 = (wv & 3) << 4;
  const int qv = wv >> 2;
  const bf16* vrow = Vt + ((size_t)b * 64 + (c0 + l16)) * NN;
  v8f acc = {};
  for (int m0 = qv * 1024; m0 < qv * 1024 + 1024; m0 += 32) {
    const float* pr = scores + l16 * RS + m0 + kbA;    // A fragment from P
    v4f p0 = *(const v4f*)(pr);
    v4f p1 = *(const v4f*)(pr + 4);
    v4f p2 = *(const v4f*)(pr + 8);
    v4f p3 = *(const v4f*)(pr + 12);
    v16bf af;
#pragma unroll
    for (int t = 0; t < 4; ++t) {
      af[t]      = (bf16)p0[t];
      af[t + 4]  = (bf16)p1[t];
      af[t + 8]  = (bf16)p2[t];
      af[t + 12] = (bf16)p3[t];
    }
    v16bf bfr = *(const v16bf*)(vrow + m0 + kbB);      // B fragment from V^T
    acc = __builtin_amdgcn_wmma_f32_16x16x32_bf16(false, af, false, bfr,
                                                  (short)0, acc, false, false);
  }
  {
    const int rb = (lane >> 4) << 3;
#pragma unroll
    for (int i = 0; i < 8; ++i)
      atomicAdd(&outAcc[(rb + i) * 64 + c0 + l16], acc[i]);  // ds_add_f32
  }
  __syncthreads();

  float* xo = xcT + (size_t)(b * NN + n0) * 64;        // token-major f32 out
  for (int idx = tid; idx < 16 * 64; idx += 512) {
    int rr = idx >> 6;
    xo[idx] = outAcc[idx] / rowsum[rr];
  }
}

// ---------------------------------------------------------------------------
// Kernel 3: gate conv (C->1, 3x3, pad 1) + sigmoid -> importance[b][n]
// ---------------------------------------------------------------------------
__global__ __launch_bounds__(256) void gate_kernel(
    const float* __restrict__ xcT, const float* __restrict__ wg,
    float* __restrict__ imp) {
  __shared__ float w[NC * 9];
  for (int i = threadIdx.x; i < NC * 9; i += 256) w[i] = wg[i];
  __syncthreads();
  int g = blockIdx.x * 256 + threadIdx.x;   // b*NN + n
  int b = g >> 12, n = g & 4095;
  int hh = n >> 6, ww = n & 63;
  float acc = 0.0f;
  for (int dy = 0; dy < 3; ++dy) {
    int y = hh + dy - 1;
    if (y < 0 || y >= NH) continue;
    for (int dx = 0; dx < 3; ++dx) {
      int xw = ww + dx - 1;
      if (xw < 0 || xw >= NW) continue;
      const float* row = xcT + ((size_t)(b * NN + y * NW + xw)) * 64;
      float s = 0.0f;
      for (int c = 0; c < NC; ++c) s = fmaf(row[c], w[c * 9 + dy * 3 + dx], s);
      acc += s;
    }
  }
  imp[g] = 1.0f / (1.0f + __expf(-acc));
}

// ---------------------------------------------------------------------------
// Kernel 4: depthwise 3x3 (pad 1) of (xc * importance) -> r1 (token-major)
// ---------------------------------------------------------------------------
__global__ __launch_bounds__(256) void dw_kernel(
    const float* __restrict__ xcT, const float* __restrict__ imp,
    const float* __restrict__ wdw, float* __restrict__ r1) {
  __shared__ float w[NC * 9];
  for (int i = threadIdx.x; i < NC * 9; i += 256) w[i] = wdw[i];
  __syncthreads();
  int g = blockIdx.x * 256 + threadIdx.x;   // (b*NN + n)*64 + c
  int c = g & 63, n = (g >> 6) & 4095, b = g >> 18;
  int hh = n >> 6, ww = n & 63;
  float acc = 0.0f;
  for (int dy = 0; dy < 3; ++dy) {
    int y = hh + dy - 1;
    if (y < 0 || y >= NH) continue;
    for (int dx = 0; dx < 3; ++dx) {
      int xw = ww + dx - 1;
      if (xw < 0 || xw >= NW) continue;
      int np = y * NW + xw;
      float gin = xcT[((size_t)(b * NN + np)) * 64 + c] * imp[b * NN + np];
      acc = fmaf(gin, w[c * 9 + dy * 3 + dx], acc);
    }
  }
  r1[g] = acc;
}

// ---------------------------------------------------------------------------
// Kernel 5: pointwise 1x1 + residual + 0.5*prev  -> xc2 (token-major)
// ---------------------------------------------------------------------------
__global__ __launch_bounds__(256) void pw_kernel(
    const float* __restrict__ xcT, const float* __restrict__ r1,
    const float* __restrict__ wpw, const float* __restrict__ prev,
    float* __restrict__ xc2) {
  __shared__ float w[NC * NC];
  for (int i = threadIdx.x; i < NC * NC; i += 256) w[i] = wpw[i];
  __syncthreads();
  int g = blockIdx.x * 256 + threadIdx.x;
  int c = g & 63, n = (g >> 6) & 4095, b = g >> 18;
  const float* rr = r1 + ((size_t)(b * NN + n)) * 64;
  float acc = 0.0f;
  for (int cp = 0; cp < NC; ++cp) acc = fmaf(rr[cp], w[c * 64 + cp], acc);
  xc2[g] = xcT[g] + acc + 0.5f * prev[((size_t)(b * NC + c)) * NN + n];
}

// ---------------------------------------------------------------------------
// Kernel 6: depthwise 3x3, dilation 2, pad 2 -> output (NCHW)
// ---------------------------------------------------------------------------
__global__ __launch_bounds__(256) void fc_kernel(
    const float* __restrict__ xc2, const float* __restrict__ wfc,
    float* __restrict__ out) {
  __shared__ float w[NC * 9];
  for (int i = threadIdx.x; i < NC * 9; i += 256) w[i] = wfc[i];
  __syncthreads();
  int g = blockIdx.x * 256 + threadIdx.x;   // ((b*64 + c)*64 + h)*64 + w
  int ww = g & 63, hh = (g >> 6) & 63, c = (g >> 12) & 63, b = g >> 18;
  float acc = 0.0f;
  for (int dy = 0; dy < 3; ++dy) {
    int y = hh + 2 * (dy - 1);
    if (y < 0 || y >= NH) continue;
    for (int dx = 0; dx < 3; ++dx) {
      int xw = ww + 2 * (dx - 1);
      if (xw < 0 || xw >= NW) continue;
      acc = fmaf(xc2[((size_t)(b * NN + y * NW + xw)) * 64 + c],
                 w[c * 9 + dy * 3 + dx], acc);
    }
  }
  out[g] = acc;
}

// ---------------------------------------------------------------------------
extern "C" void kernel_launch(void* const* d_in, const int* in_sizes, int n_in,
                              void* d_out, int out_size, void* d_ws, size_t ws_size,
                              hipStream_t stream) {
  const float* x    = (const float*)d_in[0];
  const float* prev = (const float*)d_in[1];
  const float* wqkv = (const float*)d_in[2];
  const float* wdw  = (const float*)d_in[3];
  const float* wpw  = (const float*)d_in[4];
  const float* wg   = (const float*)d_in[5];
  const float* wfc  = (const float*)d_in[6];
  float* out = (float*)d_out;

  char* ws = (char*)d_ws;                         // ~18.1 MB used
  bf16*  Qb  = (bf16*)(ws);                       // 2 MB
  bf16*  Kb  = (bf16*)(ws + (2u << 20));          // 2 MB
  bf16*  Vt  = (bf16*)(ws + (4u << 20));          // 2 MB  (V transposed)
  float* xcT = (float*)(ws + (6u << 20));         // 4 MB  attention out (token-major)
  float* imp = (float*)(ws + (10u << 20));        // 64 KB
  float* r1  = (float*)(ws + (10u << 20) + (1u << 16));   // 4 MB
  float* xc2 = (float*)(ws + (14u << 20) + (1u << 16));   // 4 MB

  qkv_kernel<<<NB * (NN / 64), 256, 0, stream>>>(x, wqkv, Qb, Kb, Vt);

  // scores 262656 + hist 16384 + qtile 2560 + outAcc 4096 + rowsum 64
  const int ATTN_LDS = 16 * RS * 4 + 16 * 256 * 4 + 16 * 80 * 2 + 16 * 64 * 4 + 64;
  (void)hipFuncSetAttribute((const void*)attn_kernel,
                            hipFuncAttributeMaxDynamicSharedMemorySize, ATTN_LDS);
  attn_kernel<<<NB * (NN / 16), 512, ATTN_LDS, stream>>>(Qb, Kb, Vt, xcT);

  gate_kernel<<<(NB * NN) / 256, 256, 0, stream>>>(xcT, wg, imp);
  dw_kernel<<<(NB * NN * NC) / 256, 256, 0, stream>>>(xcT, imp, wdw, r1);
  pw_kernel<<<(NB * NN * NC) / 256, 256, 0, stream>>>(xcT, r1, wpw, prev, xc2);
  fc_kernel<<<(NB * NC * NH * NW) / 256, 256, 0, stream>>>(xc2, wfc, out);
}
